// Net_31516470018160
// MI455X (gfx1250) — compile-verified, matrix operand order checked by reading, not run on hardware
//
#include <hip/hip_runtime.h>

// ---------------------------------------------------------------------------
// GNN TransformerConv x3 + BN + pool + MLP for MI455X (gfx1250, wave32).
// Fused q/k/v/skip projection: one wave = one 32x16 output strip (2 M-tiles),
// 8 WMMAs per 32-wide K step against 4 transposed-bf16 weight matrices.
// A fragments amortized over 4 weight matrices, B fragments over 2 M-tiles.
// Edge softmax/aggregation uses wave32 shuffles + f32 global atomics.
// ---------------------------------------------------------------------------

#define NN 20000          // nodes
#define NE 640000         // edges
#define FIN 128
#define HCC 256           // heads*channels, constant across layers
#define NG 64             // batch groups
#define BN_EPS 1e-5f

typedef __attribute__((ext_vector_type(16))) __bf16 v16bf;
typedef __attribute__((ext_vector_type(8)))  __bf16 v8bf;
typedef __attribute__((ext_vector_type(8)))  float  v8f;

// ------------------------- small utility kernels ---------------------------

__global__ void k_cvt_bf16(const float* __restrict__ s, __bf16* __restrict__ d, int n) {
  int i = blockIdx.x * blockDim.x + threadIdx.x;
  if (i < n) d[i] = (__bf16)s[i];
}

// convert f32 [K x N] row-major -> bf16 [N x K] (transposed) for contiguous
// per-column fragment loads in the WMMA GEMM.
__global__ void k_cvt_bf16_t(const float* __restrict__ s, __bf16* __restrict__ d,
                             int K, int N) {
  int i = blockIdx.x * blockDim.x + threadIdx.x;
  if (i >= K * N) return;
  int k = i / N, n = i - k * N;
  d[(size_t)n * K + k] = (__bf16)s[i];
}

__global__ void k_fill(float* __restrict__ p, float v, int n) {
  int i = blockIdx.x * blockDim.x + threadIdx.x;
  if (i < n) p[i] = v;
}

__device__ __forceinline__ void atomicMaxF(float* addr, float val) {
  unsigned* ua = reinterpret_cast<unsigned*>(addr);
  unsigned cur = *ua;
  while (__uint_as_float(cur) < val) {
    unsigned prev = atomicCAS(ua, cur, __float_as_uint(val));
    if (prev == cur) break;
    cur = prev;
  }
}

// ------------------------- fused 4-way, 2-Mtile WMMA GEMM ------------------
// Cx[M x N] = A[M x K](bf16 row-major) * Bx^T[N x K](bf16) + biasx, x=q,k,v,s
// Fragment layout (16-bit A 16x32, ISA 7.12.2): lane<16 holds row M=lane,
// K = k0 + {0..7, 16..23}; lane>=16 same row, K = k0 + {8..15, 24..31}.
// B^T gives the identical contiguous pattern per output column.
__device__ __forceinline__ v16bf load_frag(const __bf16* __restrict__ base) {
  v8bf lo = *(const v8bf*)(base);        // K chunk +0..7
  v8bf hi = *(const v8bf*)(base + 16);   // K chunk +16..23
  return __builtin_shufflevector(lo, hi, 0, 1, 2, 3, 4, 5, 6, 7,
                                 8, 9, 10, 11, 12, 13, 14, 15);
}

#define WMMA_BF16(ACC, Afrag, Bfrag) \
  (ACC) = __builtin_amdgcn_wmma_f32_16x16x32_bf16(false, (Afrag), false, (Bfrag), \
                                                  (short)0, (ACC), false, false)

__global__ void k_gemm4_wmma(const __bf16* __restrict__ A,
                             const __bf16* __restrict__ BqT, const __bf16* __restrict__ BkT,
                             const __bf16* __restrict__ BvT, const __bf16* __restrict__ BsT,
                             const float* __restrict__ bq, const float* __restrict__ bk,
                             const float* __restrict__ bv, const float* __restrict__ bs,
                             float* __restrict__ Cq, float* __restrict__ Ck,
                             float* __restrict__ Cv, float* __restrict__ Cs,
                             int M, int K, int N) {
  const int wave   = threadIdx.x >> 5;
  const int lane   = threadIdx.x & 31;
  const int half   = lane >> 4;
  const int tilesN = N >> 4;
  const int n_tile = blockIdx.y * 8 + wave;
  if (n_tile >= tilesN) return;                  // wave-uniform exit
  const int m0  = blockIdx.x * 32;               // 32-row strip = 2 M-tiles
  const int row = m0 + (lane & 15);
  const int col = n_tile * 16 + (lane & 15);

  const __bf16* arow0 = A + (size_t)row * K + half * 8;
  const __bf16* arow1 = arow0 + (size_t)16 * K;
  const __bf16* qrow  = BqT + (size_t)col * K + half * 8;
  const __bf16* krow  = BkT + (size_t)col * K + half * 8;
  const __bf16* vrow  = BvT + (size_t)col * K + half * 8;
  const __bf16* srow  = BsT + (size_t)col * K + half * 8;

  v8f aq0 = {}, ak0 = {}, av0 = {}, as0 = {};
  v8f aq1 = {}, ak1 = {}, av1 = {}, as1 = {};
  for (int k0 = 0; k0 < K; k0 += 32) {
    const v16bf a0 = load_frag(arow0 + k0);
    const v16bf a1 = load_frag(arow1 + k0);
    const v16bf fq = load_frag(qrow + k0);
    const v16bf fk = load_frag(krow + k0);
    const v16bf fv = load_frag(vrow + k0);
    const v16bf fs = load_frag(srow + k0);
    WMMA_BF16(aq0, a0, fq);  WMMA_BF16(aq1, a1, fq);
    WMMA_BF16(ak0, a0, fk);  WMMA_BF16(ak1, a1, fk);
    WMMA_BF16(av0, a0, fv);  WMMA_BF16(av1, a1, fv);
    WMMA_BF16(as0, a0, fs);  WMMA_BF16(as1, a1, fs);
  }

  const float bvq = bq[col], bvk = bk[col], bvv = bv[col], bvs = bs[col];
#pragma unroll
  for (int r = 0; r < 8; ++r) {
    const size_t rr0 = (size_t)(m0 + half * 8 + r) * N + col;       // M-tile 0
    const size_t rr1 = (size_t)(m0 + 16 + half * 8 + r) * N + col;  // M-tile 1
    Cq[rr0] = aq0[r] + bvq;  Cq[rr1] = aq1[r] + bvq;
    Ck[rr0] = ak0[r] + bvk;  Ck[rr1] = ak1[r] + bvk;
    Cv[rr0] = av0[r] + bvv;  Cv[rr1] = av1[r] + bvv;
    Cs[rr0] = as0[r] + bvs;  Cs[rr1] = as1[r] + bvs;
  }
}

// ------------------------- edge-phase kernels ------------------------------
// alpha_raw[e,h] = <q[dst], k[src] + ea@ew> / sqrt(C); segment max over dst.
// One wave per (edge, head); lanes stride channels (coalesced row reads).
__global__ void k_alpha_max(const float* __restrict__ q, const float* __restrict__ kmat,
                            const float* __restrict__ ea, const float* __restrict__ ew,
                            const int* __restrict__ src, const int* __restrict__ dst,
                            float* __restrict__ alpha, float* __restrict__ amax,
                            int H, int C, float inv_sqrt_c) {
  const int gtid = blockIdx.x * blockDim.x + threadIdx.x;
  const int wave = gtid >> 5;
  const int lane = gtid & 31;
  if (wave >= NE * H) return;                    // wave-uniform
  const int e = wave / H, h = wave - e * H;
  const int s = src[e], d = dst[e];
  const float a0 = ea[e * 4 + 0], a1 = ea[e * 4 + 1];
  const float a2 = ea[e * 4 + 2], a3 = ea[e * 4 + 3];
  const float* qrow = q    + (size_t)d * HCC + h * C;
  const float* krow = kmat + (size_t)s * HCC + h * C;
  const float* ewc  = ew + h * C;                // ew is [4, 256] row-major
  float acc = 0.f;
  for (int c = lane; c < C; c += 32) {
    const float ev = a0 * ewc[c] + a1 * ewc[256 + c] + a2 * ewc[512 + c] + a3 * ewc[768 + c];
    acc += qrow[c] * (krow[c] + ev);
  }
#pragma unroll
  for (int off = 16; off > 0; off >>= 1) acc += __shfl_xor(acc, off, 32);
  if (lane == 0) {
    const float ar = acc * inv_sqrt_c;
    alpha[(size_t)e * H + h] = ar;
    atomicMaxF(&amax[(size_t)d * H + h], ar);
  }
}

__global__ void k_alpha_expsum(float* __restrict__ alpha, const float* __restrict__ amax,
                               float* __restrict__ denom, const int* __restrict__ dst, int H) {
  const int i = blockIdx.x * blockDim.x + threadIdx.x;
  if (i >= NE * H) return;
  const int e = i / H, h = i - e * H;
  const int d = dst[e];
  const float a = __expf(alpha[i] - amax[(size_t)d * H + h]);
  alpha[i] = a;
  atomicAdd(&denom[(size_t)d * H + h], a);
}

// out[dst, t] += (v[src, t] + (ea@ew)[t]) * alpha / denom.  Block per edge,
// thread per channel (t in [0,256)); out pre-initialized with skip GEMM.
__global__ void k_message(const float* __restrict__ vmat, const float* __restrict__ ea,
                          const float* __restrict__ ew, const float* __restrict__ alpha,
                          const float* __restrict__ denom,
                          const int* __restrict__ src, const int* __restrict__ dst,
                          float* __restrict__ out, int H, int C) {
  const int e = blockIdx.x;
  const int t = threadIdx.x;
  const int h = t / C;
  const int s = src[e], d = dst[e];
  const float ev = ea[e * 4 + 0] * ew[t]       + ea[e * 4 + 1] * ew[256 + t] +
                   ea[e * 4 + 2] * ew[512 + t] + ea[e * 4 + 3] * ew[768 + t];
  const float a = alpha[(size_t)e * H + h] / (denom[(size_t)d * H + h] + 1e-16f);
  atomicAdd(&out[(size_t)d * HCC + t], (vmat[(size_t)s * HCC + t] + ev) * a);
}

// ------------------------- BatchNorm (training stats) ----------------------

__global__ void k_bn_stats(const float* __restrict__ x, float* __restrict__ sum,
                           float* __restrict__ sumsq) {
  const int c = threadIdx.x;                     // blockDim.x == 256
  float s = 0.f, s2 = 0.f;
  for (int r = blockIdx.x; r < NN; r += gridDim.x) {
    const float v = x[(size_t)r * HCC + c];
    s += v; s2 += v * v;
  }
  atomicAdd(&sum[c], s);
  atomicAdd(&sumsq[c], s2);
}

__global__ void k_bn_apply(const float* __restrict__ x, const float* __restrict__ sum,
                           const float* __restrict__ sumsq, const float* __restrict__ w,
                           const float* __restrict__ b, float* __restrict__ y,
                           __bf16* __restrict__ yb) {
  const int i = blockIdx.x * blockDim.x + threadIdx.x;
  if (i >= NN * HCC) return;
  const int c = i & (HCC - 1);
  const float mu  = sum[c]   * (1.0f / NN);
  const float var = sumsq[c] * (1.0f / NN) - mu * mu;
  const float v = (x[i] - mu) * rsqrtf(var + BN_EPS) * w[c] + b[c];
  y[i]  = v;
  yb[i] = (__bf16)v;
}

// ------------------------- pooling + MLP head ------------------------------

__global__ void k_pool(const float* __restrict__ x, const int* __restrict__ batch,
                       float* __restrict__ xadd, float* __restrict__ xmax) {
  const int i = blockIdx.x * blockDim.x + threadIdx.x;
  if (i >= NN * HCC) return;
  const int n = i / HCC, c = i - n * HCC;
  const int g = batch[n];
  const float v = x[i];
  atomicAdd(&xadd[g * HCC + c], v);
  atomicMaxF(&xmax[g * HCC + c], v);
}

__global__ void k_count(const int* __restrict__ batch, float* __restrict__ cnt) {
  const int n = blockIdx.x * blockDim.x + threadIdx.x;
  if (n < NN) atomicAdd(&cnt[batch[n]], 1.0f);
}

// h = [add | max | mean]; h1 = PReLU(h @ m1w + m1b)
__global__ void k_mlp1(const float* __restrict__ xadd, const float* __restrict__ xmax,
                       const float* __restrict__ cnt, const float* __restrict__ w,
                       const float* __restrict__ b, const float* __restrict__ pa,
                       float* __restrict__ h1) {
  const int i = blockIdx.x * blockDim.x + threadIdx.x;
  if (i >= NG * 768) return;
  const int g = i / 768, j = i - g * 768;
  const float invc = 1.0f / fmaxf(cnt[g], 1.0f);
  const float* ad = xadd + g * HCC;
  const float* mx = xmax + g * HCC;
  float acc = b[j];
  for (int c = 0; c < HCC; ++c) {
    acc += ad[c]        * w[(size_t)c * 768 + j];
    acc += mx[c]        * w[(size_t)(HCC + c) * 768 + j];
    acc += ad[c] * invc * w[(size_t)(2 * HCC + c) * 768 + j];
  }
  const float p = pa[0];
  h1[i] = acc >= 0.f ? acc : p * acc;
}

__global__ void k_mlp2(const float* __restrict__ h1, const float* __restrict__ w,
                       const float* __restrict__ b, float* __restrict__ out) {
  const int g = blockIdx.x * blockDim.x + threadIdx.x;
  if (g >= NG) return;
  float l0 = b[0], l1 = b[1];
  const float* hr = h1 + g * 768;
  for (int c = 0; c < 768; ++c) { l0 += hr[c] * w[c * 2]; l1 += hr[c] * w[c * 2 + 1]; }
  const float m = fmaxf(l0, l1);
  const float lse = m + __logf(__expf(l0 - m) + __expf(l1 - m));
  out[g * 2 + 0] = l0 - lse;
  out[g * 2 + 1] = l1 - lse;
}

// ------------------------------ orchestration ------------------------------

static inline int cdiv(long long a, int b) { return (int)((a + b - 1) / b); }

extern "C" void kernel_launch(void* const* d_in, const int* in_sizes, int n_in,
                              void* d_out, int out_size, void* d_ws, size_t ws_size,
                              hipStream_t stream) {
  (void)in_sizes; (void)n_in; (void)out_size; (void)ws_size;

  const float* x     = (const float*)d_in[0];
  const int*   ei    = (const int*)d_in[1];
  const int*   src   = ei;
  const int*   dst   = ei + NE;
  const float* ea    = (const float*)d_in[2];
  const int*   batch = (const int*)d_in[3];

  // layer parameter base: 4 + 11*l -> qw qb kw kb vw vb ew sw sb bnw bnb
  const float *QW[3], *QB[3], *KW[3], *KB[3], *VW[3], *VB[3],
              *EW[3], *SW[3], *SB[3], *BNW[3], *BNB[3];
  for (int l = 0; l < 3; ++l) {
    const int base = 4 + 11 * l;
    QW[l]  = (const float*)d_in[base + 0];  QB[l]  = (const float*)d_in[base + 1];
    KW[l]  = (const float*)d_in[base + 2];  KB[l]  = (const float*)d_in[base + 3];
    VW[l]  = (const float*)d_in[base + 4];  VB[l]  = (const float*)d_in[base + 5];
    EW[l]  = (const float*)d_in[base + 6];
    SW[l]  = (const float*)d_in[base + 7];  SB[l]  = (const float*)d_in[base + 8];
    BNW[l] = (const float*)d_in[base + 9];  BNB[l] = (const float*)d_in[base + 10];
  }
  const float* m1w = (const float*)d_in[37];
  const float* m1b = (const float*)d_in[38];
  const float* pa  = (const float*)d_in[39];
  const float* m2w = (const float*)d_in[40];
  const float* m2b = (const float*)d_in[41];
  float* out = (float*)d_out;

  // ---- carve workspace (aligned 256B) ----
  char* wsp = (char*)d_ws;
  auto carve = [&](size_t bytes) {
    char* p = wsp; wsp += (bytes + 255) & ~(size_t)255; return (void*)p;
  };
  __bf16* xb16  = (__bf16*)carve((size_t)NN * HCC * 2);
  __bf16* wqT   = (__bf16*)carve((size_t)HCC * HCC * 2);   // transposed bf16 weights
  __bf16* wkT   = (__bf16*)carve((size_t)HCC * HCC * 2);
  __bf16* wvT   = (__bf16*)carve((size_t)HCC * HCC * 2);
  __bf16* wsT   = (__bf16*)carve((size_t)HCC * HCC * 2);
  float*  qf    = (float*)carve((size_t)NN * HCC * 4);
  float*  kf    = (float*)carve((size_t)NN * HCC * 4);
  float*  vf    = (float*)carve((size_t)NN * HCC * 4);
  float*  outf  = (float*)carve((size_t)NN * HCC * 4);
  float*  xcur  = (float*)carve((size_t)NN * HCC * 4);
  float*  alpha = (float*)carve((size_t)NE * 4 * 4);
  float*  amax  = (float*)carve((size_t)NN * 4 * 4);
  float*  denom = (float*)carve((size_t)NN * 4 * 4);
  float*  bnsum = (float*)carve(HCC * 4);
  float*  bnsq  = (float*)carve(HCC * 4);
  float*  xadd  = (float*)carve((size_t)NG * HCC * 4);
  float*  xmaxb = (float*)carve((size_t)NG * HCC * 4);
  float*  cnt   = (float*)carve(NG * 4);
  float*  h1    = (float*)carve((size_t)NG * 768 * 4);

  const int TB = 256;
  const float NEG_INF = -__builtin_inff();

  // input -> bf16 (N x 128)
  k_cvt_bf16<<<cdiv((long long)NN * FIN, TB), TB, 0, stream>>>(x, xb16, NN * FIN);

  for (int l = 0; l < 3; ++l) {
    const int Kin = (l == 0) ? FIN : HCC;
    const int H   = (l == 0) ? 4 : 1;
    const int C   = HCC / H;
    const int wn  = Kin * HCC;

    // weights -> bf16, transposed to [N x K] for contiguous fragment loads
    k_cvt_bf16_t<<<cdiv(wn, TB), TB, 0, stream>>>(QW[l], wqT, Kin, HCC);
    k_cvt_bf16_t<<<cdiv(wn, TB), TB, 0, stream>>>(KW[l], wkT, Kin, HCC);
    k_cvt_bf16_t<<<cdiv(wn, TB), TB, 0, stream>>>(VW[l], wvT, Kin, HCC);
    k_cvt_bf16_t<<<cdiv(wn, TB), TB, 0, stream>>>(SW[l], wsT, Kin, HCC);

    // fused dense projections via WMMA: q, k, v, skip(->outf accumulator)
    dim3 gg(NN / 32, 2);  // 625 strips of 32 rows x (2 blocks * 8 waves = 16 N-tiles)
    k_gemm4_wmma<<<gg, TB, 0, stream>>>(xb16, wqT, wkT, wvT, wsT,
                                        QB[l], KB[l], VB[l], SB[l],
                                        qf, kf, vf, outf, NN, Kin, HCC);

    // segment softmax over dst
    k_fill<<<cdiv(NN * H, TB), TB, 0, stream>>>(amax, NEG_INF, NN * H);
    k_fill<<<cdiv(NN * H, TB), TB, 0, stream>>>(denom, 0.f, NN * H);
    k_alpha_max<<<cdiv((long long)NE * H * 32, TB), TB, 0, stream>>>(
        qf, kf, ea, EW[l], src, dst, alpha, amax, H, C, rsqrtf((float)C));
    k_alpha_expsum<<<cdiv((long long)NE * H, TB), TB, 0, stream>>>(alpha, amax, denom, dst, H);

    // message aggregation (outf already holds skip)
    k_message<<<NE, HCC, 0, stream>>>(vf, ea, EW[l], alpha, denom, src, dst, outf, H, C);

    // batchnorm -> xcur (f32) + xb16 (next layer's WMMA input)
    k_fill<<<1, TB, 0, stream>>>(bnsum, 0.f, HCC);
    k_fill<<<1, TB, 0, stream>>>(bnsq, 0.f, HCC);
    k_bn_stats<<<256, HCC, 0, stream>>>(outf, bnsum, bnsq);
    k_bn_apply<<<cdiv((long long)NN * HCC, TB), TB, 0, stream>>>(
        outf, bnsum, bnsq, BNW[l], BNB[l], xcur, xb16);
  }

  // pooling
  k_fill<<<cdiv(NG * HCC, TB), TB, 0, stream>>>(xadd, 0.f, NG * HCC);
  k_fill<<<cdiv(NG * HCC, TB), TB, 0, stream>>>(xmaxb, NEG_INF, NG * HCC);
  k_fill<<<1, NG, 0, stream>>>(cnt, 0.f, NG);
  k_pool<<<cdiv((long long)NN * HCC, TB), TB, 0, stream>>>(xcur, batch, xadd, xmaxb);
  k_count<<<cdiv(NN, TB), TB, 0, stream>>>(batch, cnt);

  // MLP head + log_softmax
  k_mlp1<<<cdiv(NG * 768, TB), TB, 0, stream>>>(xadd, xmaxb, cnt, m1w, m1b, pa, h1);
  k_mlp2<<<1, NG, 0, stream>>>(h1, m2w, m2b, out);
}
